// MultiHeadAttention_25048249270376
// MI455X (gfx1250) — matrix operand order repaired and checked by reading, not compiled, not run on hardware
//
#include <hip/hip_runtime.h>
#include <hip/hip_bf16.h>

// MHA layer for gfx1250 (wave32, WMMA f16 path).
// Compute-bound (~103 GFLOP vs ~2us of HBM traffic at 23.3 TB/s): all matmuls
// use v_wmma_f32_16x16x32_f16 with f32 accumulation.  Weight tiles for the
// dense GEMMs are staged into LDS with the Tensor Data Mover (one DMA per
// workgroup per k-step, double buffered, TENSORcnt-synchronized) so 8 waves
// share one 64x32 tile instead of each re-reading it from L2.

#define N_TOK 4096
#define DIM_  1024
#define HEADS 16
#define HD    64

typedef _Float16 half8 __attribute__((ext_vector_type(8)));
typedef _Float16 v16h  __attribute__((ext_vector_type(16)));
typedef float    v8f   __attribute__((ext_vector_type(8)));
typedef unsigned int v4u __attribute__((ext_vector_type(4)));
typedef int      v4i   __attribute__((ext_vector_type(4)));
typedef int      v8i   __attribute__((ext_vector_type(8)));

// ---- WMMA fragment helpers (CDNA5 ISA 7.12.2, wave32) ----------------------
// A 16x32 f16: lane L -> row L%16, K = 16 contiguous starting at (L>=16)?16:0 (this code's order, not the ISA table's)
static __device__ __forceinline__ v16h load_a_frag(const _Float16* p) {
  return *(const v16h*)(p);
}
// B 32x16 f16: lane L -> col L%16, K = 16 contiguous starting at (L>=16)?16:0 (this code's order, not the ISA table's)
static __device__ __forceinline__ v16h load_b_frag(const _Float16* p) {
  return *(const v16h*)(p);
}
static __device__ __forceinline__ v8f wmma_f16(v16h a, v16h b, v8f c) {
  return __builtin_amdgcn_wmma_f32_16x16x32_f16(false, a, false, b,
                                                (short)0, c, false, false);
}

// ---- Tensor Data Mover: DMA a 64(col) x 32(k) f16 tile of a row-major
// [1024][1024] matrix into LDS (ISA ch.8 D# layout).  LDS result is
// [64 cols][32 k] contiguous, exactly the B-fragment feed layout.
static __device__ __forceinline__ void tdm_load_w_tile(const _Float16* gptr,
                                                       unsigned lds_off) {
  unsigned long long ga = (unsigned long long)(uintptr_t)gptr;
  v4u g0;
  g0[0] = 1u;                                              // count=1, user mode
  g0[1] = lds_off;                                         // lds_addr (bytes)
  g0[2] = (unsigned)(ga & 0xFFFFFFFFu);                    // global_addr[31:0]
  g0[3] = (unsigned)((ga >> 32) & 0x01FFFFFFu) | (2u << 30); // ga[56:32] | type=2
  v8i g1;
  g1[0] = 0x00010000;          // workgroup_mask=0, data_size=1 (2 bytes)
  g1[1] = (int)(1024u << 16);  // tensor_dim0 = 1024 (low 16 in [31:16])
  g1[2] = (int)(1024u << 16);  // tensor_dim0 hi=0, tensor_dim1 = 1024 (low 16)
  g1[3] = (int)(32u << 16);    // tensor_dim1 hi=0, tile_dim0 = 32
  g1[4] = 64;                  // tile_dim1 = 64, tile_dim2 = 0
  g1[5] = 1024;                // tensor_dim0_stride = 1024
  g1[6] = 0;                   // stride hi, tensor_dim1_stride lo
  g1[7] = 0;
  v4i gz = {0, 0, 0, 0};       // 2D tile: groups 2/3 unused
#if __clang_major__ >= 23
  v8i gz8 = {0, 0, 0, 0, 0, 0, 0, 0};
  __builtin_amdgcn_tensor_load_to_lds(g0, g1, gz, gz, gz8, 0);
#else
  __builtin_amdgcn_tensor_load_to_lds(g0, g1, gz, gz, 0);
#endif
}

// Reductions across the 16-lane half-groups (rows of a C/D tile live per-group)
static __device__ __forceinline__ float rowmax16(float x) {
  x = fmaxf(x, __shfl_xor(x, 1, 16));
  x = fmaxf(x, __shfl_xor(x, 2, 16));
  x = fmaxf(x, __shfl_xor(x, 4, 16));
  x = fmaxf(x, __shfl_xor(x, 8, 16));
  return x;
}
static __device__ __forceinline__ float rowsum16(float x) {
  x += __shfl_xor(x, 1, 16);
  x += __shfl_xor(x, 2, 16);
  x += __shfl_xor(x, 4, 16);
  x += __shfl_xor(x, 8, 16);
  return x;
}

// ---- Prep kernels ----------------------------------------------------------
__global__ void f32_to_f16_kernel(const float* __restrict__ src,
                                  _Float16* __restrict__ dst, int n) {
  int i = blockIdx.x * blockDim.x + threadIdx.x;
  if (i < n) dst[i] = (_Float16)src[i];
}

__global__ __launch_bounds__(256) void rmsnorm_kernel(
    const float* __restrict__ emb, const float* __restrict__ w,
    _Float16* __restrict__ xh) {
  const int row = blockIdx.x, t = threadIdx.x;
  const float* e = emb + (size_t)row * DIM_;
  float s = 0.f;
  for (int k = t; k < DIM_; k += 256) { float v = e[k]; s += v * v; }
  __shared__ float red[256];
  red[t] = s; __syncthreads();
  for (int off = 128; off > 0; off >>= 1) {
    if (t < off) red[t] += red[t + off];
    __syncthreads();
  }
  const float inv = rsqrtf(red[0] / (float)DIM_ + 1e-6f);
  for (int k = t; k < DIM_; k += 256)
    xh[(size_t)row * DIM_ + k] = (_Float16)(e[k] * inv * w[k]);
}

// sequence_ids is sorted: allowed mask per row i is the dense prefix
// j < max(i+1, end_of_run(seq_id[i])).  Binary search per token.
__global__ void jmax_kernel(const int* __restrict__ seq, int* __restrict__ jmax,
                            int n) {
  int i = blockIdx.x * blockDim.x + threadIdx.x;
  if (i >= n) return;
  int id = seq[i];
  int lo = i, hiE = n;
  while (lo < hiE) { int mid = (lo + hiE) >> 1; if (seq[mid] <= id) lo = mid + 1; else hiE = mid; }
  int e = lo;                      // first index with larger seq id
  jmax[i] = (i + 1 > e) ? (i + 1) : e;
}

// ---- QKV projection GEMM: Y = clip(x @ W^T) --------------------------------
// Block = 8 waves = 128 rows x 64 cols (one head).  Weight tile (64x32) is
// TDM-staged into double-buffered LDS, shared by all 8 waves.
// mode 0/1: RoPE then store [HEADS][N][64]   mode 2: store transposed [HEADS][64][N]
__global__ __launch_bounds__(256) void qkv_gemm_kernel(
    const _Float16* __restrict__ xh, const _Float16* __restrict__ Wh,
    const int* __restrict__ gidx,
    _Float16* __restrict__ outRope, _Float16* __restrict__ outTrans, int mode) {
  const int tid  = threadIdx.x;
  const int lane = tid & 31;
  const int wv   = tid >> 5;            // wave in block, 0..7
  const int rb   = blockIdx.x & 31;     // row block (128 rows each)
  const int head = blockIdx.x >> 5;     // 0..15 (column group of 64)
  const int i0 = rb * 128 + wv * 16;
  const int col0 = head * 64;
  const int n  = lane & 15;
  const int hi = lane >> 4;

  __shared__ __align__(128) _Float16 Bl[2][64 * 32];   // [buf][col][k]
  const unsigned lds_base = (unsigned)(uintptr_t)&Bl[0][0];

  const _Float16* arow  = xh + (size_t)(i0 + n) * DIM_ + hi * 16;
  const _Float16* wtile = Wh + (size_t)col0 * DIM_;    // + k0 per step

  if (tid < 32) tdm_load_w_tile(wtile, lds_base);      // prologue: buf 0

  v8f acc0 = {}, acc1 = {}, acc2 = {}, acc3 = {};
  int p = 0;
  for (int k0 = 0; k0 < DIM_; k0 += 32) {
    __builtin_amdgcn_s_wait_tensorcnt(0);   // own-wave TDM complete (no-op if none)
    __syncthreads();                        // buffer p published to all waves
    if (tid < 32 && (k0 + 32) < DIM_)
      tdm_load_w_tile(wtile + k0 + 32, lds_base + (unsigned)((p ^ 1) * 64 * 32 * 2));
    if (k0 + 128 < DIM_) __builtin_prefetch(arow + k0 + 128, 0, 0);
    v16h a = load_a_frag(arow + k0);
    const _Float16* bl = &Bl[p][(size_t)n * 32 + hi * 16];
    acc0 = wmma_f16(a, load_b_frag(bl),           acc0);
    acc1 = wmma_f16(a, load_b_frag(bl + 16 * 32), acc1);
    acc2 = wmma_f16(a, load_b_frag(bl + 32 * 32), acc2);
    acc3 = wmma_f16(a, load_b_frag(bl + 48 * 32), acc3);
    __syncthreads();                        // all done reading buf p
    p ^= 1;
  }

  v8f acc[4] = {acc0, acc1, acc2, acc3};
#pragma unroll
  for (int t = 0; t < 4; t++)
#pragma unroll
    for (int v = 0; v < 8; v++)
      acc[t][v] = fminf(fmaxf(acc[t][v], -8.0f), 8.0f);

  if (mode == 2) {                       // V: store transposed [h][d][token]
#pragma unroll
    for (int t = 0; t < 4; t++)
#pragma unroll
      for (int v = 0; v < 8; v++) {
        int row = i0 + v + hi * 8;
        outTrans[((size_t)head * HD + t * 16 + n) * N_TOK + row] = (_Float16)acc[t][v];
      }
  } else {                               // Q/K: RoPE (rotate-half), theta=5e5
    const float kNegL2T = -18.931568569324174f / 32.0f;  // -log2(5e5)/32
#pragma unroll
    for (int tp = 0; tp < 2; tp++) {
      const float invf = exp2f((float)(tp * 16 + n) * kNegL2T); // dd&31 same for tp,tp+2
#pragma unroll
      for (int v = 0; v < 8; v++) {
        int row = i0 + v + hi * 8;
        float ang = (float)gidx[row] * invf;
        float s, c; __sincosf(ang, &s, &c);
        float x1 = acc[tp][v], x2 = acc[tp + 2][v];
        size_t base = ((size_t)head * N_TOK + row) * HD + n;
        outRope[base + tp * 16]       = (_Float16)(x1 * c - x2 * s);
        outRope[base + (tp + 2) * 16] = (_Float16)(x2 * c + x1 * s);
      }
    }
  }
}

// ---- Flash attention: one wave per (head, 16-token block), j in steps of 32
__global__ __launch_bounds__(32) void attn_kernel(
    const _Float16* __restrict__ Qh, const _Float16* __restrict__ Kh,
    const _Float16* __restrict__ Vth, const int* __restrict__ jmax,
    _Float16* __restrict__ Ah) {
  const int lane = threadIdx.x & 31;
  const int n = lane & 15, hi = lane >> 4;
  const int i0 = blockIdx.x * 16;
  const int head = blockIdx.y;
  __shared__ __align__(32) _Float16 P[16 * 32];  // f32 C-layout -> f16 A-layout

  const _Float16* qrow = Qh + ((size_t)head * N_TOK + i0 + n) * HD + hi * 16;
  v16h aq0 = load_a_frag(qrow);        // d = 0..31
  v16h aq1 = load_a_frag(qrow + 32);   // d = 32..63

  float mi[8], li[8]; int jm[8];
  v8f o0 = {}, o1 = {}, o2 = {}, o3 = {};
#pragma unroll
  for (int v = 0; v < 8; v++) { mi[v] = -1e30f; li[v] = 0.f; jm[v] = jmax[i0 + v + hi * 8]; }
  const int jend = jmax[i0 + 15];      // jmax is non-decreasing

  const _Float16* kbase  = Kh  + (size_t)head * N_TOK * HD + hi * 16;
  const _Float16* vbase0 = Vth + (size_t)head * HD * N_TOK + hi * 16;

  for (int j0 = 0; j0 < jend; j0 += 32) {
    const _Float16* kp0 = kbase + (size_t)(j0 + n) * HD;
    const _Float16* kp1 = kbase + (size_t)(j0 + 16 + n) * HD;
    v8f s0 = {}, s1 = {};
    s0 = wmma_f16(aq0, load_b_frag(kp0), s0);
    s0 = wmma_f16(aq1, load_b_frag(kp0 + 32), s0);
    s1 = wmma_f16(aq0, load_b_frag(kp1), s1);
    s1 = wmma_f16(aq1, load_b_frag(kp1 + 32), s1);

#pragma unroll
    for (int v = 0; v < 8; v++) {
      float a0 = s0[v] * 0.125f; if (j0 + n      >= jm[v]) a0 = -1e30f;
      float a1 = s1[v] * 0.125f; if (j0 + 16 + n >= jm[v]) a1 = -1e30f;
      float r  = rowmax16(fmaxf(a0, a1));
      float nm = fmaxf(mi[v], r);
      float alpha = __expf(mi[v] - nm);
      float p0 = __expf(a0 - nm), p1 = __expf(a1 - nm);
      float rs = rowsum16(p0 + p1);
      li[v] = li[v] * alpha + rs;
      mi[v] = nm;
      o0[v] *= alpha; o1[v] *= alpha; o2[v] *= alpha; o3[v] *= alpha;
      int m = v + hi * 8;
      P[m * 32 + n]      = (_Float16)p0;
      P[m * 32 + 16 + n] = (_Float16)p1;
    }
    __syncthreads();                       // single-wave WG: NOP + DS wait
    v16h ap = load_a_frag(&P[n * 32 + hi * 16]);
    const _Float16* vb = vbase0 + j0;
    o0 = wmma_f16(ap, load_b_frag(vb + (size_t)(n)      * N_TOK), o0);
    o1 = wmma_f16(ap, load_b_frag(vb + (size_t)(16 + n) * N_TOK), o1);
    o2 = wmma_f16(ap, load_b_frag(vb + (size_t)(32 + n) * N_TOK), o2);
    o3 = wmma_f16(ap, load_b_frag(vb + (size_t)(48 + n) * N_TOK), o3);
    __syncthreads();
  }
#pragma unroll
  for (int v = 0; v < 8; v++) {
    float inv = 1.0f / li[v];
    int row = i0 + v + hi * 8;
    size_t base = (size_t)row * DIM_ + head * HD + n;
    Ah[base]      = (_Float16)(o0[v] * inv);
    Ah[base + 16] = (_Float16)(o1[v] * inv);
    Ah[base + 32] = (_Float16)(o2[v] * inv);
    Ah[base + 48] = (_Float16)(o3[v] * inv);
  }
}

// ---- Output GEMM: out(f32) = attn @ Wo^T, TDM-staged weight tile -----------
__global__ __launch_bounds__(256) void out_gemm_kernel(
    const _Float16* __restrict__ Ah, const _Float16* __restrict__ Woh,
    float* __restrict__ out) {
  const int tid  = threadIdx.x;
  const int lane = tid & 31;
  const int wv   = tid >> 5;
  const int rb   = blockIdx.x & 31;
  const int cg   = blockIdx.x >> 5;
  const int i0 = rb * 128 + wv * 16;
  const int col0 = cg * 64;
  const int n  = lane & 15;
  const int hi = lane >> 4;

  __shared__ __align__(128) _Float16 Bl[2][64 * 32];
  const unsigned lds_base = (unsigned)(uintptr_t)&Bl[0][0];

  const _Float16* arow  = Ah  + (size_t)(i0 + n) * DIM_ + hi * 16;
  const _Float16* wtile = Woh + (size_t)col0 * DIM_;

  if (tid < 32) tdm_load_w_tile(wtile, lds_base);

  v8f acc0 = {}, acc1 = {}, acc2 = {}, acc3 = {};
  int p = 0;
  for (int k0 = 0; k0 < DIM_; k0 += 32) {
    __builtin_amdgcn_s_wait_tensorcnt(0);
    __syncthreads();
    if (tid < 32 && (k0 + 32) < DIM_)
      tdm_load_w_tile(wtile + k0 + 32, lds_base + (unsigned)((p ^ 1) * 64 * 32 * 2));
    if (k0 + 128 < DIM_) __builtin_prefetch(arow + k0 + 128, 0, 0);
    v16h a = load_a_frag(arow + k0);
    const _Float16* bl = &Bl[p][(size_t)n * 32 + hi * 16];
    acc0 = wmma_f16(a, load_b_frag(bl),           acc0);
    acc1 = wmma_f16(a, load_b_frag(bl + 16 * 32), acc1);
    acc2 = wmma_f16(a, load_b_frag(bl + 32 * 32), acc2);
    acc3 = wmma_f16(a, load_b_frag(bl + 48 * 32), acc3);
    __syncthreads();
    p ^= 1;
  }
  v8f acc[4] = {acc0, acc1, acc2, acc3};
#pragma unroll
  for (int t = 0; t < 4; t++)
#pragma unroll
    for (int v = 0; v < 8; v++) {
      int row = i0 + v + hi * 8;
      out[(size_t)row * DIM_ + col0 + t * 16 + n] = acc[t][v];
    }
}

// ---- Host launch -----------------------------------------------------------
extern "C" void kernel_launch(void* const* d_in, const int* in_sizes, int n_in,
                              void* d_out, int out_size, void* d_ws, size_t ws_size,
                              hipStream_t stream) {
  const float* emb = (const float*)d_in[0];
  const float* nw  = (const float*)d_in[1];
  const float* Wq  = (const float*)d_in[2];
  const float* Wk  = (const float*)d_in[3];
  const float* Wv  = (const float*)d_in[4];
  const float* Wo  = (const float*)d_in[5];
  const int* gidx  = (const int*)d_in[7];
  const int* seqid = (const int*)d_in[8];
  // d_in[6] sequence_indexes and d_in[9] mask_pad are unused by the reference math.

  char* w = (char*)d_ws;
  auto alloc = [&](size_t bytes) -> char* {
    char* p = w; w += (bytes + 255) & ~(size_t)255; return p;
  };
  _Float16* xh  = (_Float16*)alloc((size_t)N_TOK * DIM_ * 2);
  _Float16* Wqh = (_Float16*)alloc((size_t)DIM_ * DIM_ * 2);
  _Float16* Wkh = (_Float16*)alloc((size_t)DIM_ * DIM_ * 2);
  _Float16* Wvh = (_Float16*)alloc((size_t)DIM_ * DIM_ * 2);
  _Float16* Woh = (_Float16*)alloc((size_t)DIM_ * DIM_ * 2);
  _Float16* Qh  = (_Float16*)alloc((size_t)HEADS * N_TOK * HD * 2);
  _Float16* Kh  = (_Float16*)alloc((size_t)HEADS * N_TOK * HD * 2);
  _Float16* Vth = (_Float16*)alloc((size_t)HEADS * HD * N_TOK * 2);
  _Float16* Ah  = (_Float16*)alloc((size_t)N_TOK * DIM_ * 2);
  int* jmaxb    = (int*)alloc((size_t)N_TOK * 4);

  const int WN = DIM_ * DIM_;
  f32_to_f16_kernel<<<(WN + 255) / 256, 256, 0, stream>>>(Wq, Wqh, WN);
  f32_to_f16_kernel<<<(WN + 255) / 256, 256, 0, stream>>>(Wk, Wkh, WN);
  f32_to_f16_kernel<<<(WN + 255) / 256, 256, 0, stream>>>(Wv, Wvh, WN);
  f32_to_f16_kernel<<<(WN + 255) / 256, 256, 0, stream>>>(Wo, Woh, WN);

  rmsnorm_kernel<<<N_TOK, 256, 0, stream>>>(emb, nw, xh);
  jmax_kernel<<<N_TOK / 256, 256, 0, stream>>>(seqid, jmaxb, N_TOK);

  // 512 blocks: 32 row-blocks (128 rows) x 16 heads, 8 waves/block
  qkv_gemm_kernel<<<512, 256, 0, stream>>>(xh, Wqh, gidx, Qh, nullptr, 0);
  qkv_gemm_kernel<<<512, 256, 0, stream>>>(xh, Wkh, gidx, Kh, nullptr, 1);
  qkv_gemm_kernel<<<512, 256, 0, stream>>>(xh, Wvh, gidx, nullptr, Vth, 2);

  dim3 ag(N_TOK / 16, HEADS);
  attn_kernel<<<ag, 32, 0, stream>>>(Qh, Kh, Vth, jmaxb, Ah);

  out_gemm_kernel<<<512, 256, 0, stream>>>(Ah, Woh, (float*)d_out);
}
